// GCBlock_12532714569875
// MI455X (gfx1250) — matrix operand order repaired and checked by reading, not compiled, not run on hardware
//
#include <hip/hip_runtime.h>
#include <cmath>

typedef __attribute__((ext_vector_type(16))) _Float16 v16h;
typedef __attribute__((ext_vector_type(2)))  _Float16 h2v;
typedef __attribute__((ext_vector_type(2)))  __fp16   pk2;   // cvt_pkrtz result type
typedef __attribute__((ext_vector_type(8)))  float    v8f;

#define ASTRIDE 70   // halfs per activation row; bank stride 35 (coprime 64) -> conflict-free A reads

#if __has_builtin(__builtin_amdgcn_tanhf)
#define TANHF(x) __builtin_amdgcn_tanhf(x)
#else
#define TANHF(x) tanhf(x)
#endif

// ---------------- helpers ----------------

__device__ __forceinline__ void zero4(v8f c[4]) {
#pragma unroll
  for (int nt = 0; nt < 4; ++nt)
#pragma unroll
    for (int i = 0; i < 8; ++i) c[nt][i] = 0.0f;
}

// One 16x64 (A, LDS f16 tile) @ 64x64 (B, LDS pre-swizzled f16) -> 16x64 f32 (4 C frags)
// using v_wmma_f32_16x16x32_f16. kc-outer/nt-inner ordering keeps 3 independent
// WMMAs between each RAW-dependent pair (f16 WMMA->WMMA hazard window).
__device__ __forceinline__ void wave_gemm16x64(
    const _Float16* __restrict__ act,   // LDS: 16 rows x ASTRIDE halfs
    const _Float16* __restrict__ wsw,   // LDS: 8 frags x 32 lanes x 16 halfs (f = kc*4+nt)
    unsigned lane, v8f c[4])
{
  const unsigned row = lane & 15u;
  const unsigned kb  = (lane >> 4) << 4;          // 0 or 16 (A-layout K base)
  v16h afr[2];
#pragma unroll
  for (int kc = 0; kc < 2; ++kc) {
#pragma unroll
    for (int v = 0; v < 8; ++v) {
      // A in contiguous k order (B packed to match): lanes0-15 K{0..15}, lanes16-31 K{16..31}
      const unsigned k = (unsigned)kc * 32u + kb + 2u * v;
      h2v p = *(const h2v*)(act + row * ASTRIDE + k);
      afr[kc][2 * v]     = p.x;
      afr[kc][2 * v + 1] = p.y;
    }
  }
#pragma unroll
  for (int kc = 0; kc < 2; ++kc) {
#pragma unroll
    for (int nt = 0; nt < 4; ++nt) {
      const v16h b = *(const v16h*)(wsw + (((unsigned)(kc * 4 + nt) * 32u + lane) << 4));
      c[nt] = __builtin_amdgcn_wmma_f32_16x16x32_f16(false, afr[kc], false, b,
                                                     (short)0, c[nt], false, false);
    }
  }
}

// bias + (optional) tanh, write f16 activations back to the wave's LDS tile.
// C layout: VGPR r -> M = r (+8 for lanes 16-31), N = lane&15 within N-tile.
template <bool TANH>
__device__ __forceinline__ void bias_act_store(
    const v8f c[4], const float* __restrict__ bias,
    _Float16* __restrict__ act, unsigned lane)
{
  const unsigned n_l = lane & 15u;
  const unsigned hi  = lane >> 4;
#pragma unroll
  for (int nt = 0; nt < 4; ++nt) {
    const float b = bias[nt * 16 + n_l];
#pragma unroll
    for (int r = 0; r < 8; ++r) {
      float v = c[nt][r] + b;
      if (TANH) v = TANHF(v);
      act[(r + hi * 8) * ASTRIDE + nt * 16 + n_l] = (_Float16)v;
    }
  }
}

// Convert f32 weights [64x64, k-major] into WMMA B-fragment order in LDS:
// frag f = kc*4+nt ; lane L holds column N = nt*16+(L&15),
// K = kc*32 + (L>=16 ? 16:0) + e  for e = 0..15 (contiguous 32B per lane).
__device__ __forceinline__ void swizzle_weights(
    const float* W0, const float* W1, const float* W2, const float* W3,
    int nlayers, _Float16* wswz, unsigned t, unsigned nthreads)
{
  const unsigned total = (unsigned)nlayers << 8;   // layers * 8 frags * 32 lanes
  for (unsigned idx = t; idx < total; idx += nthreads) {
    const unsigned l   = idx >> 8;
    const unsigned rem = idx & 255u;
    const unsigned f   = rem >> 5;
    const unsigned L   = rem & 31u;
    const unsigned kc  = f >> 2, nt = f & 3u;
    const unsigned n     = nt * 16u + (L & 15u);
    const unsigned kbase = kc * 32u + ((L >> 4) << 4);
    const float* W = (l == 0) ? W0 : (l == 1) ? W1 : (l == 2) ? W2 : W3;
    _Float16* dst = wswz + ((size_t)idx << 4);
#pragma unroll
    for (int e = 0; e < 16; ++e)
      dst[e] = (_Float16)W[(kbase + (unsigned)e) * 64u + n];
  }
}

// ---------------- kernels ----------------

__global__ void zero_kernel(float4* __restrict__ out, unsigned n4) {
  unsigned i = blockIdx.x * blockDim.x + threadIdx.x;
  if (i < n4) out[i] = make_float4(0.f, 0.f, 0.f, 0.f);
}

// pp1 = MLP(p1) over nodes; output f32 to workspace.
__global__ __launch_bounds__(256) void node_mlp_kernel(
    const float* __restrict__ p1,
    const float* __restrict__ w1, const float* __restrict__ w2,
    const float* __restrict__ b1, const float* __restrict__ b2,
    float* __restrict__ pp1, int n_tiles)
{
  __shared__ alignas(32) _Float16 wswz[2 * 8 * 32 * 16];   // 16 KB
  __shared__ float bias_s[2 * 64];
  __shared__ alignas(16) _Float16 act[8][16 * ASTRIDE];    // per-wave tiles

  const unsigned t = threadIdx.x;
  swizzle_weights(w1, w2, w1, w1, 2, wswz, t, blockDim.x);
  if (t < 128) bias_s[t] = (t < 64) ? b1[t] : b2[t - 64];
  __syncthreads();

  const unsigned lane = t & 31u, wid = t >> 5;
  _Float16* myact = &act[wid][0];
  const unsigned gw = blockIdx.x * 8u + wid;
  const unsigned nw = gridDim.x * 8u;
  const unsigned c0 = lane * 2u;
  const unsigned n_l = lane & 15u, hi = lane >> 4;

  for (unsigned tile = gw; tile < (unsigned)n_tiles; tile += nw) {
    const unsigned node0 = tile * 16u;
#pragma unroll 4
    for (int r = 0; r < 16; ++r) {
      const float2 a = *(const float2*)(p1 + ((node0 + (unsigned)r) << 6) + c0);
      *(pk2*)(myact + r * ASTRIDE + c0) = __builtin_amdgcn_cvt_pkrtz(a.x, a.y);
    }
    v8f c[4];
    zero4(c);
    wave_gemm16x64(myact, wswz, lane, c);
    bias_act_store<true>(c, bias_s, myact, lane);
    zero4(c);
    wave_gemm16x64(myact, wswz + 8 * 32 * 16, lane, c);
#pragma unroll
    for (int r = 0; r < 8; ++r) {
      const unsigned node = node0 + (unsigned)r + hi * 8u;
#pragma unroll
      for (int nt = 0; nt < 4; ++nt)
        pp1[(node << 6) + nt * 16 + n_l] = c[nt][r] + bias_s[64 + nt * 16 + n_l];
    }
  }
}

// Fused edge pipeline: gather -> pi MLP -> ii MLP -> atomic scatter-add.
__global__ __launch_bounds__(256) void edge_kernel(
    const int* __restrict__ idx_i, const int* __restrict__ idx_j,
    const float* __restrict__ basis, const float* __restrict__ pp1,
    const float* __restrict__ w0, const float* __restrict__ w1,
    const float* __restrict__ w2, const float* __restrict__ w3,
    const float* __restrict__ b0, const float* __restrict__ b1,
    const float* __restrict__ b2, const float* __restrict__ b3,
    float* __restrict__ out, int n_tiles)
{
  __shared__ alignas(32) _Float16 wswz[4 * 8 * 32 * 16];   // 32 KB
  __shared__ float bias_s[4 * 64];
  __shared__ alignas(16) _Float16 act[8][16 * ASTRIDE];

  const unsigned t = threadIdx.x;
  swizzle_weights(w0, w1, w2, w3, 4, wswz, t, blockDim.x);
  {
    const unsigned l = t >> 6, pos = t & 63u;
    const float* B = (l == 0) ? b0 : (l == 1) ? b1 : (l == 2) ? b2 : b3;
    bias_s[t] = B[pos];
  }
  __syncthreads();

  const unsigned lane = t & 31u, wid = t >> 5;
  _Float16* myact = &act[wid][0];
  const unsigned gw = blockIdx.x * 8u + wid;
  const unsigned nw = gridDim.x * 8u;
  const unsigned c0 = lane * 2u;
  const unsigned n_l = lane & 15u, hi = lane >> 4;

  for (unsigned tile = gw; tile < (unsigned)n_tiles; tile += nw) {
    const unsigned e0 = tile * 16u;
    // gather: inter = pp1[idx_i] + pp1[idx_j] + basis   (f32 add, packed f16 store)
#pragma unroll 4
    for (int r = 0; r < 16; ++r) {
      const unsigned i = (unsigned)idx_i[e0 + (unsigned)r];   // wave-uniform
      const unsigned j = (unsigned)idx_j[e0 + (unsigned)r];
      const float2 a  = *(const float2*)(pp1   + (i << 6) + c0);
      const float2 b  = *(const float2*)(pp1   + (j << 6) + c0);
      const float2 bb = *(const float2*)(basis + ((e0 + (unsigned)r) << 6) + c0);
      *(pk2*)(myact + r * ASTRIDE + c0) =
          __builtin_amdgcn_cvt_pkrtz(a.x + b.x + bb.x, a.y + b.y + bb.y);
    }
    v8f c[4];
    zero4(c);                                            // pi layer 1 (+tanh)
    wave_gemm16x64(myact, wswz + 0 * 4096, lane, c);
    bias_act_store<true>(c, bias_s + 0, myact, lane);
    zero4(c);                                            // pi layer 2
    wave_gemm16x64(myact, wswz + 1 * 4096, lane, c);
    bias_act_store<false>(c, bias_s + 64, myact, lane);
    zero4(c);                                            // ii layer 1 (+tanh)
    wave_gemm16x64(myact, wswz + 2 * 4096, lane, c);
    bias_act_store<true>(c, bias_s + 128, myact, lane);
    zero4(c);                                            // ii layer 2
    wave_gemm16x64(myact, wswz + 3 * 4096, lane, c);
    // scatter-add ii1 rows to out[idx_i]
#pragma unroll
    for (int r = 0; r < 8; ++r) {
      const unsigned m = (unsigned)r + hi * 8u;
      const unsigned i = (unsigned)idx_i[e0 + m];
#pragma unroll
      for (int nt = 0; nt < 4; ++nt) {
        const float v = c[nt][r] + bias_s[192 + nt * 16 + n_l];
        unsafeAtomicAdd(out + (i << 6) + nt * 16 + n_l, v);
      }
    }
  }
}

// ---------------- launch ----------------

extern "C" void kernel_launch(void* const* d_in, const int* in_sizes, int n_in,
                              void* d_out, int out_size, void* d_ws, size_t ws_size,
                              hipStream_t stream) {
  const int*   idx_i = (const int*)d_in[0];
  const int*   idx_j = (const int*)d_in[1];
  const float* p1    = (const float*)d_in[2];
  const float* basis = (const float*)d_in[3];
  const float* pp_w1 = (const float*)d_in[4];
  const float* pp_w2 = (const float*)d_in[5];
  const float* pi_w1 = (const float*)d_in[6];
  const float* pi_w2 = (const float*)d_in[7];
  const float* ii_w1 = (const float*)d_in[8];
  const float* ii_w2 = (const float*)d_in[9];
  const float* pp_b1 = (const float*)d_in[10];
  const float* pp_b2 = (const float*)d_in[11];
  const float* pi_b1 = (const float*)d_in[12];
  const float* pi_b2 = (const float*)d_in[13];
  const float* ii_b1 = (const float*)d_in[14];
  const float* ii_b2 = (const float*)d_in[15];

  float* out = (float*)d_out;
  float* pp1 = (float*)d_ws;                 // 100000*64*4 = 25.6 MB scratch

  const int N = 100000;
  const int E = 1200000;

  const unsigned n4 = (unsigned)(N * 64 / 4);
  zero_kernel<<<(n4 + 255) / 256, 256, 0, stream>>>((float4*)out, n4);

  node_mlp_kernel<<<782, 256, 0, stream>>>(p1, pp_w1, pp_w2, pp_b1, pp_b2,
                                           pp1, N / 16);

  edge_kernel<<<2048, 256, 0, stream>>>(idx_i, idx_j, basis, pp1,
                                        pi_w1, pi_w2, ii_w1, ii_w2,
                                        pi_b1, pi_b2, ii_b1, ii_b2,
                                        out, E / 16);
}
